// NonLocal_88141318848736
// MI455X (gfx1250) — compile-verified, matrix operand order checked by reading, not run on hardware
//
#include <hip/hip_runtime.h>
#include <hip/hip_bf16.h>

// ---------------------------------------------------------------------------
// Types for CDNA5 WMMA / TDM (gfx1250, wave32)
// ---------------------------------------------------------------------------
typedef __attribute__((ext_vector_type(16))) __bf16 v16bf;
typedef __attribute__((ext_vector_type(8)))  float  v8f;
typedef unsigned int u32x4 __attribute__((ext_vector_type(4)));
typedef int          i32x8 __attribute__((ext_vector_type(8)));
typedef int          i32x4 __attribute__((ext_vector_type(4)));

union frag16 {
    uint4 u[2];   // 32 bytes
    v16bf v;
};

__device__ __forceinline__ unsigned short f32_to_bf16(float f) {
    unsigned int bits = __builtin_bit_cast(unsigned int, f);
    bits += 0x7FFFu + ((bits >> 16) & 1u);   // round-to-nearest-even
    return (unsigned short)(bits >> 16);
}

// ---------------------------------------------------------------------------
// TDM: async DMA of one contiguous 8 KB weight tile (4096 bf16) into LDS.
// D# built per CDNA5 ISA ch.8: group0 = {count/type, lds_addr, global_addr},
// group1 = {data_size=2B, tensor_dim0=tile_dim0=4096, 1 row}.
// Issued by one wave; tracked with TENSORcnt.
// ---------------------------------------------------------------------------
__device__ __forceinline__ void tdm_load_8kb(unsigned lds_addr, const void* gptr) {
    unsigned long long ga = (unsigned long long)(uintptr_t)gptr;
    u32x4 g0;
    g0[0] = 1u;                                            // count=1 (valid descriptor)
    g0[1] = lds_addr;                                      // LDS byte address
    g0[2] = (unsigned)(ga & 0xFFFFFFFFu);                  // global_addr[31:0]
    g0[3] = (unsigned)((ga >> 32) & 0x1FFFFFFu)            // global_addr[56:32]
          | (2u << 30);                                    // type = 2 ("image")
    i32x8 g1;
    g1[0] = (int)(1u << 16);        // data_size = 1 -> 2 bytes/elem
    g1[1] = (int)(4096u << 16);     // tensor_dim0[15:0] = 4096
    g1[2] = (int)(1u << 16);        // tensor_dim0[31:16]=0 | tensor_dim1 = 1
    g1[3] = (int)(4096u << 16);     // tile_dim0 = 4096
    g1[4] = 1;                      // tile_dim1 = 1, tile_dim2 = 0
    g1[5] = 4096;                   // tensor_dim0_stride[31:0]
    g1[6] = 0;
    g1[7] = 0;
    i32x4 z4 = {0, 0, 0, 0};
#if defined(__clang_major__) && (__clang_major__ >= 23)
    i32x8 z8 = {0, 0, 0, 0, 0, 0, 0, 0};
    __builtin_amdgcn_tensor_load_to_lds(g0, g1, z4, z4, z8, 0);
#else
    __builtin_amdgcn_tensor_load_to_lds(g0, g1, z4, z4, 0);
#endif
}

// ---------------------------------------------------------------------------
// Kernel 0: concat f_w/g_w/h_w -> bf16 weight matrix wcat[512][256]
// ---------------------------------------------------------------------------
__global__ __launch_bounds__(256) void prep_weights(
    const float* __restrict__ f_w, const float* __restrict__ g_w,
    const float* __restrict__ h_w, unsigned short* __restrict__ wcat)
{
    int i = blockIdx.x * 256 + threadIdx.x;      // 512*256 = 131072 total
    if (i >= 512 * 256) return;
    int m = i >> 8, c = i & 255;
    float v = (m < 128) ? f_w[m * 256 + c]
            : (m < 256) ? g_w[(m - 128) * 256 + c]
                        : h_w[(m - 256) * 256 + c];
    wcat[i] = f32_to_bf16(v);
}

// ---------------------------------------------------------------------------
// Kernel 1: fused conv1x1 (f,g,h stacked as M=512) + 16x16 maxpool, via WMMA.
// Grid: B*256 blocks, one per (batch, pooled tile). Block: 256 threads (8 waves).
// Each wave owns 2 spatial rows (32 pixels); its bf16 x-fragments (B matrices)
// live in VGPRs for the whole M loop. Weights double-buffered via TDM async DMA.
// pooled layout: [B][512][256]  (channel-major, tile index t = ti*16+tj)
// ---------------------------------------------------------------------------
__global__ __launch_bounds__(256) void fused_conv_pool(
    const float* __restrict__ x, const unsigned short* __restrict__ wcat,
    const float* __restrict__ f_b, const float* __restrict__ g_b,
    const float* __restrict__ h_b, float* __restrict__ pooled)
{
    __shared__ __align__(16) unsigned short wbuf[2][16][256];  // 16 KB double-buffered weights
    __shared__ __align__(16) unsigned short xstage[8][32][40]; // 20 KB per-wave transpose buffers
    __shared__ float pmax[8][16];                              // per-wave channel maxima

    const int tid  = threadIdx.x;
    const int wv   = tid >> 5;
    const int lane = tid & 31;
    const int blk  = blockIdx.x;
    const int b    = blk >> 8;
    const int t    = blk & 255;          // pooled tile index
    const int ti   = t >> 4, tj = t & 15;

    // this lane's pixel within the wave's 2 rows
    const int prow = lane >> 4;          // 0 or 1
    const int pcol = lane & 15;
    const int gy   = ti * 16 + wv * 2 + prow;
    const int gx   = tj * 16 + pcol;
    const float* xp = x + (((size_t)b * 256) * 256 + gy) * 256 + gx;

    // kick off TDM load of M-tile 0 into buffer 0 (wave 0 only)
    const unsigned lds_wbuf = (unsigned)(uintptr_t)(void*)&wbuf[0][0][0];
    if (tid < 32) tdm_load_8kb(lds_wbuf, wcat);

    // -------- Phase A: build B fragments (x tile, bf16) in registers --------
    // B layout (32x16 K x N, 16-bit): lanes 0-15 N=0..15 hold K {0..7,16..23},
    // lanes 16-31 hold K {8..15,24..31} for the same N.
    frag16 Bf[2][8];
    const int ko1 = (lane < 16) ? 0 : 8;
    for (int kc = 0; kc < 8; ++kc) {
        // coalesced: 32 channels for this lane's own pixel -> LDS [pixel][chan]
        #pragma unroll
        for (int cc = 0; cc < 32; ++cc) {
            float v = xp[(size_t)(kc * 32 + cc) * 65536];
            xstage[wv][lane][cc] = f32_to_bf16(v);
        }
        // wave-internal transpose read (LDS ops are in-order per wave)
        #pragma unroll
        for (int n = 0; n < 2; ++n) {
            const int p = n * 16 + pcol;
            const uint4* src = (const uint4*)&xstage[wv][p][0];
            Bf[n][kc].u[0] = src[ko1 >> 3];          // K ko1 .. ko1+7
            Bf[n][kc].u[1] = src[(ko1 + 16) >> 3];   // K ko1+16 .. ko1+23
        }
    }

    // -------- Phase B: loop over 32 M-tiles of 16 output channels --------
    const int mrow = lane & 15;
    for (int mt = 0; mt < 32; ++mt) {
        const int cur = mt & 1;
        // wave 0: wait for this tile's DMA, publish to all waves
        if (tid < 32) __builtin_amdgcn_s_wait_tensorcnt(0);
        __syncthreads();
        // wave 0: prefetch next tile into the other buffer (safe: the barrier
        // above proves everyone finished reading it in the previous iteration)
        if (tid < 32 && (mt + 1) < 32)
            tdm_load_8kb(lds_wbuf + 8192u * (unsigned)(1 - cur),
                         wcat + (size_t)(mt + 1) * 4096);

        v8f acc0 = {}; v8f acc1 = {};
        #pragma unroll
        for (int kc = 0; kc < 8; ++kc) {
            // A layout (16x32 M x K, 16-bit): lane holds row M=lane%16,
            // lanes 0-15 K {0..7,16..23}, lanes 16-31 K {8..15,24..31}
            frag16 A;
            const uint4* ws = (const uint4*)&wbuf[cur][mrow][kc * 32];
            A.u[0] = ws[ko1 >> 3];
            A.u[1] = ws[(ko1 + 16) >> 3];
            acc0 = __builtin_amdgcn_wmma_f32_16x16x32_bf16(
                false, A.v, false, Bf[0][kc].v, (short)0, acc0, false, false);
            acc1 = __builtin_amdgcn_wmma_f32_16x16x32_bf16(
                false, A.v, false, Bf[1][kc].v, (short)0, acc1, false, false);
        }

        // max over this wave's 32 pixels: elementwise then across the 16 N-lanes
        v8f mx;
        #pragma unroll
        for (int j = 0; j < 8; ++j) mx[j] = fmaxf(acc0[j], acc1[j]);
        #pragma unroll
        for (int off = 1; off < 16; off <<= 1) {
            #pragma unroll
            for (int j = 0; j < 8; ++j)
                mx[j] = fmaxf(mx[j], __shfl_xor(mx[j], off, 32));
        }
        // D layout: lanes 0-15 hold M = j, lanes 16-31 hold M = j+8
        if ((lane & 15) == 0) {
            const int half = lane >> 4;
            #pragma unroll
            for (int j = 0; j < 8; ++j)
                pmax[wv][half * 8 + j] = mx[j];
        }
        __syncthreads();

        // combine the 8 wave-partials, add bias, write pooled result
        if (tid < 16) {
            float v = pmax[0][tid];
            #pragma unroll
            for (int w = 1; w < 8; ++w) v = fmaxf(v, pmax[w][tid]);
            const int chan = mt * 16 + tid;
            const float bias = (chan < 128) ? f_b[chan]
                             : (chan < 256) ? g_b[chan - 128]
                                            : h_b[chan - 256];
            pooled[((size_t)b * 512 + chan) * 256 + t] = v + bias;
        }
        __syncthreads();
    }
}

// ---------------------------------------------------------------------------
// Kernel 2: attention over pooled positions (tiny: ~0.2 GFLOP total).
// One block per batch; thread q handles query position q.
// pooled = [fv(128) | gv(128) | hv(256)] each [chan][256].
// ---------------------------------------------------------------------------
__global__ __launch_bounds__(256) void attention(
    const float* __restrict__ pooled, float* __restrict__ S,
    float* __restrict__ att)
{
    const int b = blockIdx.x;
    const int q = threadIdx.x;
    const float* fv = pooled + (size_t)b * 512 * 256;
    const float* gv = fv + 128 * 256;
    const float* hv = fv + 256 * 256;

    float fq[128];
    #pragma unroll
    for (int c = 0; c < 128; ++c) fq[c] = fv[c * 256 + q];

    float* Srow = S + ((size_t)b * 256 + q) * 256;
    float m = -3.0e38f;
    for (int k = 0; k < 256; ++k) {
        float s = 0.f;
        #pragma unroll
        for (int c = 0; c < 128; ++c) s += fq[c] * gv[c * 256 + k];
        Srow[k] = s;
        m = fmaxf(m, s);
    }
    float sum = 0.f;
    for (int k = 0; k < 256; ++k) {
        float p = __expf(Srow[k] - m);
        Srow[k] = p;
        sum += p;
    }
    const float inv = 1.f / sum;
    for (int k = 0; k < 256; ++k) Srow[k] *= inv;

    for (int c = 0; c < 256; ++c) {
        float acc = 0.f;
        for (int k = 0; k < 256; ++k) acc += Srow[k] * hv[c * 256 + k];
        att[((size_t)b * 256 + c) * 256 + q] = acc;   // [B][C][i*16+j]
    }
}

// ---------------------------------------------------------------------------
// Kernel 3: bilinear upsample 16x16 -> 256x256 (half-pixel) + residual add.
// Pure bandwidth: float4 (b128) loads/stores; 4 pixels share the y-lerp row.
// ---------------------------------------------------------------------------
__global__ __launch_bounds__(256) void upsample_add(
    const float* __restrict__ att, const float* __restrict__ xin,
    float* __restrict__ out)
{
    const size_t i4 = (size_t)blockIdx.x * 256 + threadIdx.x;  // 16,777,216 total
    const int xq = (int)(i4 & 63);            // group of 4 pixels along W
    const int y  = (int)((i4 >> 6) & 255);
    const int c  = (int)((i4 >> 14) & 255);
    const int b  = (int)(i4 >> 22);
    const float* a = att + ((size_t)b * 256 + c) * 256;

    float sy = (y + 0.5f) * (1.0f / 16.0f) - 0.5f;
    sy = fminf(fmaxf(sy, 0.0f), 15.0f);
    const int y0 = (int)sy;
    const int y1 = min(y0 + 1, 15);
    const float wy = sy - (float)y0;

    const float4 xv = ((const float4*)xin)[i4];
    const float xvp[4] = {xv.x, xv.y, xv.z, xv.w};
    float rp[4];
    #pragma unroll
    for (int j = 0; j < 4; ++j) {
        const int xc = xq * 4 + j;
        float sx = (xc + 0.5f) * (1.0f / 16.0f) - 0.5f;
        sx = fminf(fmaxf(sx, 0.0f), 15.0f);
        const int x0 = (int)sx;
        const int x1 = min(x0 + 1, 15);
        const float wx = sx - (float)x0;
        const float v0 = a[y0 * 16 + x0] * (1.f - wx) + a[y0 * 16 + x1] * wx;
        const float v1 = a[y1 * 16 + x0] * (1.f - wx) + a[y1 * 16 + x1] * wx;
        rp[j] = v0 * (1.f - wy) + v1 * wy + xvp[j];
    }
    ((float4*)out)[i4] = make_float4(rp[0], rp[1], rp[2], rp[3]);
}

// ---------------------------------------------------------------------------
// Launch
// ---------------------------------------------------------------------------
extern "C" void kernel_launch(void* const* d_in, const int* in_sizes, int n_in,
                              void* d_out, int out_size, void* d_ws, size_t ws_size,
                              hipStream_t stream) {
    const float* x   = (const float*)d_in[0];
    const float* f_w = (const float*)d_in[1];
    const float* f_b = (const float*)d_in[2];
    const float* g_w = (const float*)d_in[3];
    const float* g_b = (const float*)d_in[4];
    const float* h_w = (const float*)d_in[5];
    const float* h_b = (const float*)d_in[6];
    float* out = (float*)d_out;

    // workspace layout
    char* ws = (char*)d_ws;
    unsigned short* wcat = (unsigned short*)ws;              // 512*256 bf16 = 256 KB
    size_t off = 512 * 256 * sizeof(unsigned short);
    float* pooled = (float*)(ws + off);                      // 4*512*256 f32 = 2 MB
    off += (size_t)4 * 512 * 256 * sizeof(float);
    float* S = (float*)(ws + off);                           // 4*256*256 f32 = 1 MB
    off += (size_t)4 * 256 * 256 * sizeof(float);
    float* att = (float*)(ws + off);                         // 4*256*256 f32 = 1 MB

    prep_weights<<<512, 256, 0, stream>>>(f_w, g_w, h_w, wcat);
    fused_conv_pool<<<4 * 256, 256, 0, stream>>>(x, wcat, f_b, g_b, h_b, pooled);
    attention<<<4, 256, 0, stream>>>(pooled, S, att);
    upsample_add<<<(4u * 256u * 256u * 256u) / 4u / 256u, 256, 0, stream>>>(att, x, out);
}
